// MeshEGNN_40123584479325
// MI455X (gfx1250) — matrix operand order repaired and checked by reading, not compiled, not run on hardware
//
#include <hip/hip_runtime.h>
#include <hip/hip_bf16.h>
#include <math.h>

#define NNODES 1024
#define HDIM 64
#define EHID 258
#define EHID_PAD 288
#define MDIM 16
#define WAVES 4   // waves per block in edge kernels (1 node i per wave)

typedef __bf16 bf16_t;
typedef __attribute__((ext_vector_type(16))) __bf16 v16bf;
typedef __attribute__((ext_vector_type(8)))  __bf16 v8bf;
typedef __attribute__((ext_vector_type(8)))  float  v8f;

__device__ __forceinline__ float silu_f(float x) { return x / (1.f + __expf(-x)); }

// Within-wave LDS producer->consumer ordering: wait all outstanding DS ops.
__device__ __forceinline__ void wave_lds_sync() {
  __builtin_amdgcn_wave_barrier();
  asm volatile("s_wait_dscnt 0" ::: "memory");
  __builtin_amdgcn_wave_barrier();
}

// A-operand (16x32 bf16 tile of a [16 rows][ldk] LDS buffer), per ISA layout:
// lanes 0-15 hold row=lane, K = {0..7, 16..23}; lanes 16-31: K = {8..15, 24..31}.
__device__ __forceinline__ v16bf lds_amat(const bf16_t* base, int ldk, int kOff) {
  const int lane = threadIdx.x & 31;
  const bf16_t* p = base + (lane & 15) * ldk + kOff + ((lane >> 4) << 3);
  v8bf lo = *(const v8bf*)p;
  v8bf hi = *(const v8bf*)(p + 16);
  v16bf r;
#pragma unroll
  for (int t = 0; t < 8; ++t) { r[t] = lo[t]; r[t + 8] = hi[t]; }
  return r;
}

// B-operand (32x16 bf16 tile). LDS buffer is stored column-major: [col][ldk K].
// lane = column (lane&15), lanes 0-15 hold K = {0..7, 16..23}; lanes 16-31: K = {8..15, 24..31}.
__device__ __forceinline__ v16bf lds_bmat(const bf16_t* base, int ldk, int kOff, int colOff) {
  const int lane = threadIdx.x & 31;
  const bf16_t* p = base + (colOff + (lane & 15)) * ldk + kOff + ((lane >> 4) << 3);
  v8bf lo = *(const v8bf*)p;
  v8bf hi = *(const v8bf*)(p + 16);
  v16bf r;
#pragma unroll
  for (int t = 0; t < 8; ++t) { r[t] = lo[t]; r[t + 8] = hi[t]; }
  return r;
}

__device__ __forceinline__ v8f wmma_bf16f32(v16bf a, v16bf b, v8f c) {
  return __builtin_amdgcn_wmma_f32_16x16x32_bf16(false, a, false, b, (short)0, c, false, false);
}

// ---------------------------------------------------------------------------
// Generic small dense layer: Y = act(X @ W + b) (+resid). O(N) node-level work.
// act: 0 none, 1 silu, 2 relu, 3 sigmoid.  transOut: write Y[n*M+m].
// ---------------------------------------------------------------------------
__global__ void dense_kernel(const float* __restrict__ X, const float* __restrict__ W,
                             const float* __restrict__ bias, const float* __restrict__ resid,
                             float* __restrict__ Y, int M, int K, int Nc, int Wstride,
                             int act, int transOut) {
  int idx = blockIdx.x * blockDim.x + threadIdx.x;
  if (idx >= M * Nc) return;
  int m = idx / Nc, n = idx - m * Nc;
  float s = bias ? bias[n] : 0.f;
  const float* xr = X + m * K;
  for (int k = 0; k < K; ++k) s = fmaf(xr[k], W[k * Wstride + n], s);
  if (act == 1) s = silu_f(s);
  else if (act == 2) s = fmaxf(s, 0.f);
  else if (act == 3) s = 1.f / (1.f + __expf(-s));
  if (resid) s += resid[m * Nc + n];
  Y[transOut ? (n * M + m) : idx] = s;
}

__global__ void concat_kernel(const float* __restrict__ h, const float* __restrict__ mi,
                              float* __restrict__ hc) {
  int idx = blockIdx.x * blockDim.x + threadIdx.x;
  if (idx >= NNODES * (HDIM + MDIM)) return;
  int m = idx / (HDIM + MDIM), k = idx - m * (HDIM + MDIM);
  hc[idx] = (k < HDIM) ? h[m * HDIM + k] : mi[m * MDIM + (k - HDIM)];
}

__global__ void combine_kernel(const float* __restrict__ feats, const float* __restrict__ coors,
                               const float* __restrict__ gam, const float* __restrict__ nbr,
                               const float* __restrict__ nnb, const float* __restrict__ Cs,
                               float* __restrict__ out_vel, float* __restrict__ out_coors) {
  int idx = blockIdx.x * blockDim.x + threadIdx.x;
  if (idx >= NNODES * 3) return;
  int i = idx / 3, c = idx - i * 3;
  float denom = fmaxf(nnb[i], 1.f);
  float vel = gam[i] * feats[i * 8 + 1 + c] + Cs[0] * nbr[idx] / denom;
  out_vel[idx] = vel;
  out_coors[idx] = coors[idx] + vel;
}

// ---------------------------------------------------------------------------
// EGNN layer edge kernel (fused, WMMA).  Per wave: one node i, sweep j tiles.
//   pre(i,j,:) = A[i,:] + BmT[:,j] + d2*w1d  (first edge GEMM factored out)
//   m = silu(silu(pre) @ W2 + b2) * mask        -> sum_j into m_i[i]
//   cw = (silu(m @ cw1 + cb1) . cw2 + cb2)*mask -> x_out[i] = x[i] + sum cw*rel
// ---------------------------------------------------------------------------
__global__ __launch_bounds__(WAVES * 32)
void egnn_edge_layer_kernel(const float* __restrict__ x_in,
                            const float* __restrict__ Amat,  // [N][258], b1 folded in
                            const float* __restrict__ BmT,   // [258][N]
                            const float* __restrict__ w1d,   // [258] (d2 row of w1)
                            const int*   __restrict__ adj,   // [N][N]
                            const float* __restrict__ w2,    // [258][16]
                            const float* __restrict__ b2,    // [16]
                            const float* __restrict__ cw1,   // [16][64]
                            const float* __restrict__ cb1,   // [64]
                            const float* __restrict__ cw2,   // [64]
                            const float* __restrict__ cb2,   // [1]
                            float* __restrict__ m_i,         // [N][16]
                            float* __restrict__ x_out) {     // [N][3]
  extern __shared__ char smem[];
  bf16_t* W2T   = (bf16_t*)smem;                       // [16 cols][288 K]
  bf16_t* CW1B  = W2T + 16 * EHID_PAD;                 // [64 cols][32 K]
  bf16_t* Ebase = CW1B + 64 * 32;                      // WAVES * [16][288]
  bf16_t* Tbase = Ebase + WAVES * 16 * EHID_PAD;       // WAVES * [16][32]
  float*  cwbuf = (float*)(Tbase + WAVES * 16 * 32);   // WAVES * 16

  const int tid = threadIdx.x;
  for (int idx = tid; idx < 16 * EHID_PAD; idx += blockDim.x) {
    int n = idx / EHID_PAD, k = idx - n * EHID_PAD;
    W2T[idx] = (bf16_t)((k < EHID) ? w2[k * MDIM + n] : 0.f);
  }
  for (int idx = tid; idx < 64 * 32; idx += blockDim.x) {
    int n = idx >> 5, k = idx & 31;
    CW1B[idx] = (bf16_t)((k < MDIM) ? cw1[k * 64 + n] : 0.f);
  }
  __syncthreads();

  const int lane = tid & 31;
  const int wv   = tid >> 5;
  const int i    = blockIdx.x * WAVES + wv;

  bf16_t* Ebuf = Ebase + wv * 16 * EHID_PAD;
  bf16_t* Tbuf = Tbase + wv * 16 * 32;
  float*  cwb  = cwbuf + wv * 16;

  // zero static pad regions once
  for (int idx = lane; idx < 16 * 32; idx += 32)
    Ebuf[(idx >> 5) * EHID_PAD + 256 + (idx & 31)] = (bf16_t)0.f;
  for (int idx = lane; idx < 16 * 16; idx += 32)
    Tbuf[(idx >> 4) * 32 + 16 + (idx & 15)] = (bf16_t)0.f;

  const float xi0 = x_in[i * 3 + 0], xi1 = x_in[i * 3 + 1], xi2 = x_in[i * 3 + 2];
  const float* Ai = Amat + i * EHID;
  const float b2v = b2[lane & 15];
  float cb1v[4], cw2v[4];
#pragma unroll
  for (int nt = 0; nt < 4; ++nt) {
    cb1v[nt] = cb1[nt * 16 + (lane & 15)];
    cw2v[nt] = cw2[nt * 16 + (lane & 15)];
  }
  const float cb2s = cb2[0];
  const int rowOff = (lane >> 4) << 3;

  float miacc = 0.f, xacc0 = 0.f, xacc1 = 0.f, xacc2 = 0.f;

  for (int jb = 0; jb < NNODES; jb += 16) {
    const int jj = jb + (lane & 15);
    const float r0 = xi0 - x_in[jj * 3 + 0];
    const float r1 = xi1 - x_in[jj * 3 + 1];
    const float r2 = xi2 - x_in[jj * 3 + 2];
    const float d2 = r0 * r0 + r1 * r1 + r2 * r2;
    const float maskv = ((adj[i * NNODES + jj] != 0) || (jj == i)) ? 1.f : 0.f;

    // Build silu(pre) tile: rows = j-in-tile, 2 features per iteration across halves.
    {
      bf16_t* erow = Ebuf + (lane & 15) * EHID_PAD;
      for (int f = (lane >> 4); f < EHID; f += 2) {
        float v = Ai[f] + BmT[f * NNODES + jj] + d2 * w1d[f];
        erow[f] = (bf16_t)silu_f(v);
      }
    }
    wave_lds_sync();

    // m2 = E @ W2 : K=288 -> 9 WMMAs
    v8f acc = {};
#pragma unroll
    for (int kc = 0; kc < 9; ++kc) {
      v16bf a = lds_amat(Ebuf, EHID_PAD, kc * 32);
      v16bf b = lds_bmat(W2T, EHID_PAD, kc * 32, 0);
      acc = wmma_bf16f32(a, b, acc);
    }

    float mrow[8];
    float part = 0.f;
#pragma unroll
    for (int r = 0; r < 8; ++r) {
      mrow[r] = __shfl(maskv, r + rowOff, 32);
      float c = silu_f(acc[r] + b2v) * mrow[r];
      part += c;
      Tbuf[(rowOff + r) * 32 + (lane & 15)] = (bf16_t)c;  // transpose m -> A layout
    }
    part += __shfl_xor(part, 16, 32);
    miacc += part;
    wave_lds_sync();

    // coors MLP: t = silu(m @ cw1 + cb1); cw = t . cw2 + cb2
    v16bf am = lds_amat(Tbuf, 32, 0);
    float dot[8] = {0.f, 0.f, 0.f, 0.f, 0.f, 0.f, 0.f, 0.f};
#pragma unroll
    for (int nt = 0; nt < 4; ++nt) {
      v16bf bw = lds_bmat(CW1B, 32, 0, nt * 16);
      v8f z = {};
      v8f a2 = wmma_bf16f32(am, bw, z);
#pragma unroll
      for (int r = 0; r < 8; ++r) dot[r] += silu_f(a2[r] + cb1v[nt]) * cw2v[nt];
    }
#pragma unroll
    for (int r = 0; r < 8; ++r) {
      dot[r] += __shfl_xor(dot[r], 1, 32);
      dot[r] += __shfl_xor(dot[r], 2, 32);
      dot[r] += __shfl_xor(dot[r], 4, 32);
      dot[r] += __shfl_xor(dot[r], 8, 32);
      dot[r] = (dot[r] + cb2s) * mrow[r];
    }
    if (lane == 0) {
#pragma unroll
      for (int r = 0; r < 8; ++r) cwb[r] = dot[r];
    }
    if (lane == 16) {
#pragma unroll
      for (int r = 0; r < 8; ++r) cwb[8 + r] = dot[r];
    }
    wave_lds_sync();
    if (lane < 16) {
      float cwj = cwb[lane];
      xacc0 = fmaf(cwj, r0, xacc0);
      xacc1 = fmaf(cwj, r1, xacc1);
      xacc2 = fmaf(cwj, r2, xacc2);
    }
    wave_lds_sync();  // protect cwb before next tile overwrites
  }

  if (lane < 16) m_i[i * MDIM + lane] = miacc;
  xacc0 += __shfl_xor(xacc0, 1, 32); xacc0 += __shfl_xor(xacc0, 2, 32);
  xacc0 += __shfl_xor(xacc0, 4, 32); xacc0 += __shfl_xor(xacc0, 8, 32);
  xacc1 += __shfl_xor(xacc1, 1, 32); xacc1 += __shfl_xor(xacc1, 2, 32);
  xacc1 += __shfl_xor(xacc1, 4, 32); xacc1 += __shfl_xor(xacc1, 8, 32);
  xacc2 += __shfl_xor(xacc2, 1, 32); xacc2 += __shfl_xor(xacc2, 2, 32);
  xacc2 += __shfl_xor(xacc2, 4, 32); xacc2 += __shfl_xor(xacc2, 8, 32);
  if (lane == 0) {
    x_out[i * 3 + 0] = xi0 + xacc0;
    x_out[i * 3 + 1] = xi1 + xacc1;
    x_out[i * 3 + 2] = xi2 + xacc2;
  }
}

// ---------------------------------------------------------------------------
// Final stage edge kernel: phi_e (factored) + phi_x, masked neighbor aggregation.
// ---------------------------------------------------------------------------
__global__ __launch_bounds__(WAVES * 32)
void egnn_final_edge_kernel(const float* __restrict__ coors,
                            const float* __restrict__ P,    // [N][64], eb1 folded
                            const float* __restrict__ QT,   // [64][N]
                            const float* __restrict__ ed,   // [64]
                            const int*   __restrict__ adj,
                            const float* __restrict__ ew2,  // [64][64]
                            const float* __restrict__ eb2,  // [64]
                            const float* __restrict__ xw1,  // [64][64]
                            const float* __restrict__ xb1,  // [64]
                            const float* __restrict__ xw2,  // [64]
                            const float* __restrict__ xb2,  // [1]
                            float* __restrict__ nbr,        // [N][3]
                            float* __restrict__ nnb) {      // [N]
  extern __shared__ char smem[];
  bf16_t* EW2T  = (bf16_t*)smem;                      // [64 cols][64 K]
  bf16_t* XW1T  = EW2T + 64 * 64;                     // [64 cols][64 K]
  bf16_t* Ebase = XW1T + 64 * 64;                     // WAVES * [16][64]
  bf16_t* Tbase = Ebase + WAVES * 16 * 64;            // WAVES * [16][64]
  float*  cwbuf = (float*)(Tbase + WAVES * 16 * 64);  // WAVES * 16

  const int tid = threadIdx.x;
  for (int idx = tid; idx < 64 * 64; idx += blockDim.x) {
    int n = idx >> 6, k = idx & 63;
    EW2T[idx] = (bf16_t)ew2[k * 64 + n];
    XW1T[idx] = (bf16_t)xw1[k * 64 + n];
  }
  __syncthreads();

  const int lane = tid & 31;
  const int wv   = tid >> 5;
  const int i    = blockIdx.x * WAVES + wv;

  bf16_t* Ebuf = Ebase + wv * 16 * 64;
  bf16_t* Tbuf = Tbase + wv * 16 * 64;
  float*  cwb  = cwbuf + wv * 16;

  const float xi0 = coors[i * 3 + 0], xi1 = coors[i * 3 + 1], xi2 = coors[i * 3 + 2];
  const float* Pi = P + i * HDIM;
  float eb2v[4], xb1v[4], xw2v[4];
#pragma unroll
  for (int nt = 0; nt < 4; ++nt) {
    eb2v[nt] = eb2[nt * 16 + (lane & 15)];
    xb1v[nt] = xb1[nt * 16 + (lane & 15)];
    xw2v[nt] = xw2[nt * 16 + (lane & 15)];
  }
  const float xb2s = xb2[0];
  const int rowOff = (lane >> 4) << 3;

  float n0 = 0.f, n1 = 0.f, n2 = 0.f, cnt = 0.f;

  for (int jb = 0; jb < NNODES; jb += 16) {
    const int jj = jb + (lane & 15);
    const float r0 = xi0 - coors[jj * 3 + 0];
    const float r1 = xi1 - coors[jj * 3 + 1];
    const float r2 = xi2 - coors[jj * 3 + 2];
    const float d2 = r0 * r0 + r1 * r1 + r2 * r2;
    const float m2v = ((adj[i * NNODES + jj] != 0) && (jj != i)) ? 1.f : 0.f;

    {
      bf16_t* erow = Ebuf + (lane & 15) * 64;
      for (int f = (lane >> 4); f < HDIM; f += 2) {
        float v = Pi[f] + QT[f * NNODES + jj] + d2 * ed[f];
        erow[f] = (bf16_t)fmaxf(v, 0.f);
      }
    }
    wave_lds_sync();

    // m_ij = relu(E @ ew2 + eb2): 4 col tiles x 2 K chunks
#pragma unroll
    for (int nt = 0; nt < 4; ++nt) {
      v8f acc = {};
#pragma unroll
      for (int kc = 0; kc < 2; ++kc) {
        v16bf a = lds_amat(Ebuf, 64, kc * 32);
        v16bf b = lds_bmat(EW2T, 64, kc * 32, nt * 16);
        acc = wmma_bf16f32(a, b, acc);
      }
#pragma unroll
      for (int r = 0; r < 8; ++r) {
        float mv = fmaxf(acc[r] + eb2v[nt], 0.f);
        Tbuf[(rowOff + r) * 64 + nt * 16 + (lane & 15)] = (bf16_t)mv;
      }
    }
    wave_lds_sync();

    // phi = tanh(relu(m_ij @ xw1 + xb1) . xw2 + xb2)
    float dot[8] = {0.f, 0.f, 0.f, 0.f, 0.f, 0.f, 0.f, 0.f};
#pragma unroll
    for (int nt = 0; nt < 4; ++nt) {
      v8f acc = {};
#pragma unroll
      for (int kc = 0; kc < 2; ++kc) {
        v16bf a = lds_amat(Tbuf, 64, kc * 32);
        v16bf b = lds_bmat(XW1T, 64, kc * 32, nt * 16);
        acc = wmma_bf16f32(a, b, acc);
      }
#pragma unroll
      for (int r = 0; r < 8; ++r) dot[r] += fmaxf(acc[r] + xb1v[nt], 0.f) * xw2v[nt];
    }
#pragma unroll
    for (int r = 0; r < 8; ++r) {
      dot[r] += __shfl_xor(dot[r], 1, 32);
      dot[r] += __shfl_xor(dot[r], 2, 32);
      dot[r] += __shfl_xor(dot[r], 4, 32);
      dot[r] += __shfl_xor(dot[r], 8, 32);
      float mrow = __shfl(m2v, r + rowOff, 32);
      dot[r] = tanhf(dot[r] + xb2s) * mrow;
    }
    if (lane == 0) {
#pragma unroll
      for (int r = 0; r < 8; ++r) cwb[r] = dot[r];
    }
    if (lane == 16) {
#pragma unroll
      for (int r = 0; r < 8; ++r) cwb[8 + r] = dot[r];
    }
    wave_lds_sync();
    if (lane < 16) {
      float w = cwb[lane];
      n0 = fmaf(w, r0, n0);
      n1 = fmaf(w, r1, n1);
      n2 = fmaf(w, r2, n2);
      cnt += m2v;
    }
    wave_lds_sync();
  }

  n0 += __shfl_xor(n0, 1, 32); n0 += __shfl_xor(n0, 2, 32);
  n0 += __shfl_xor(n0, 4, 32); n0 += __shfl_xor(n0, 8, 32);
  n1 += __shfl_xor(n1, 1, 32); n1 += __shfl_xor(n1, 2, 32);
  n1 += __shfl_xor(n1, 4, 32); n1 += __shfl_xor(n1, 8, 32);
  n2 += __shfl_xor(n2, 1, 32); n2 += __shfl_xor(n2, 2, 32);
  n2 += __shfl_xor(n2, 4, 32); n2 += __shfl_xor(n2, 8, 32);
  cnt += __shfl_xor(cnt, 1, 32); cnt += __shfl_xor(cnt, 2, 32);
  cnt += __shfl_xor(cnt, 4, 32); cnt += __shfl_xor(cnt, 8, 32);
  if (lane == 0) {
    nbr[i * 3 + 0] = n0;
    nbr[i * 3 + 1] = n1;
    nbr[i * 3 + 2] = n2;
    nnb[i] = cnt;
  }
}

// ---------------------------------------------------------------------------
// Host orchestration
// ---------------------------------------------------------------------------
static void launch_dense(const float* X, const float* W, const float* bias, const float* resid,
                         float* Y, int M, int K, int Nc, int Wstride, int act, int transOut,
                         hipStream_t s) {
  int total = M * Nc;
  dense_kernel<<<(total + 255) / 256, 256, 0, s>>>(X, W, bias, resid, Y, M, K, Nc, Wstride,
                                                   act, transOut);
}

extern "C" void kernel_launch(void* const* d_in, const int* in_sizes, int n_in,
                              void* d_out, int out_size, void* d_ws, size_t ws_size,
                              hipStream_t stream) {
  (void)n_in; (void)ws_size;
  // ---- resolve flatten order: insertion-order dict walk vs sorted pytree ----
  int I_feats, I_coors, I_adj, I_Win, I_bin, I_C;
  int Lw1[2], Lb1[2], Lw2[2], Lb2[2], Lcw1[2], Lcb1[2], Lcw2[2], Lcb2[2];
  int Lnw1[2], Lnb1[2], Lnw2[2], Lnb2[2];
  int I_ew1, I_eb1, I_ew2, I_eb2, I_vw1, I_vb1, I_vw2, I_vb2;
  int I_xw1, I_xb1, I_xw2, I_xb2, I_sw, I_sb;
  if (in_sizes[0] == NNODES * 8) {
    // insertion order: feats, coors, adj, input_mlp, layers(edge,coors,node), phi_v, phi_e, phi_x, stress, C
    I_feats = 0; I_coors = 1; I_adj = 2; I_Win = 3; I_bin = 4;
    int b = 5;
    for (int l = 0; l < 2; ++l) {
      Lw1[l] = b + 0;  Lb1[l] = b + 1;  Lw2[l] = b + 2;  Lb2[l] = b + 3;
      Lcw1[l] = b + 4; Lcb1[l] = b + 5; Lcw2[l] = b + 6; Lcb2[l] = b + 7;
      Lnw1[l] = b + 8; Lnb1[l] = b + 9; Lnw2[l] = b + 10; Lnb2[l] = b + 11;
      b += 12;
    }
    I_vw1 = b + 0; I_vb1 = b + 1; I_vw2 = b + 2; I_vb2 = b + 3;
    I_ew1 = b + 4; I_eb1 = b + 5; I_ew2 = b + 6; I_eb2 = b + 7;
    I_xw1 = b + 8; I_xb1 = b + 9; I_xw2 = b + 10; I_xb2 = b + 11;
    I_sw = b + 12; I_sb = b + 13; I_C = b + 14;
  } else {
    // sorted pytree order: adj, coors, feats, C, input_mlp, layers(coors,edge,node), phi_e, phi_v, phi_x, stress
    I_adj = 0; I_coors = 1; I_feats = 2; I_C = 3; I_Win = 4; I_bin = 5;
    int b = 6;
    for (int l = 0; l < 2; ++l) {
      Lcw1[l] = b + 0; Lcb1[l] = b + 1; Lcw2[l] = b + 2; Lcb2[l] = b + 3;
      Lw1[l] = b + 4;  Lb1[l] = b + 5;  Lw2[l] = b + 6;  Lb2[l] = b + 7;
      Lnw1[l] = b + 8; Lnb1[l] = b + 9; Lnw2[l] = b + 10; Lnb2[l] = b + 11;
      b += 12;
    }
    I_ew1 = b + 0; I_eb1 = b + 1; I_ew2 = b + 2; I_eb2 = b + 3;
    I_vw1 = b + 4; I_vb1 = b + 5; I_vw2 = b + 6; I_vb2 = b + 7;
    I_xw1 = b + 8; I_xb1 = b + 9; I_xw2 = b + 10; I_xb2 = b + 11;
    I_sw = b + 12; I_sb = b + 13;
  }
  #define FIN(i) ((const float*)d_in[i])
  const float* feats = FIN(I_feats);
  const float* coors = FIN(I_coors);
  const int*   adj   = (const int*)d_in[I_adj];

  // ---- workspace carve-out (floats) ----
  float* w = (float*)d_ws;
  float* h    = w;             w += NNODES * HDIM;       // 65536
  float* hc   = w;             w += NNODES * 80;         // 81920
  float* Amat = w;             w += NNODES * EHID;       // 264192
  float* BmT  = w;             w += EHID * NNODES;       // 264192
  float* mi   = w;             w += NNODES * MDIM;       // 16384
  float* t1   = w;             w += NNODES * 128;        // 131072
  float* xA   = w;             w += NNODES * 3;
  float* xB   = w;             w += NNODES * 3;
  float* Pbuf = w;             w += NNODES * HDIM;
  float* QT   = w;             w += HDIM * NNODES;
  float* nbr  = w;             w += NNODES * 3;
  float* nnb  = w;             w += NNODES;
  float* tv   = w;             w += NNODES * HDIM;
  float* gbuf = w;             w += NNODES;

  float* out_vel    = (float*)d_out;                // [N*3]
  float* out_stress = out_vel + NNODES * 3;         // [N]
  float* out_coors  = out_stress + NNODES;          // [N*3]
  (void)out_size;

  const size_t lds1 = (16 * EHID_PAD + 64 * 32 + WAVES * 16 * EHID_PAD + WAVES * 16 * 32) * sizeof(bf16_t)
                      + WAVES * 16 * sizeof(float);
  const size_t lds2 = (64 * 64 * 2 + WAVES * 16 * 64 * 2) * sizeof(bf16_t)
                      + WAVES * 16 * sizeof(float);

  // h0 = feats @ Win + bin
  launch_dense(feats, FIN(I_Win), FIN(I_bin), nullptr, h, NNODES, 8, HDIM, HDIM, 0, 0, stream);

  const float* xin = coors;
  float* xouts[2] = {xA, xB};
  for (int l = 0; l < 2; ++l) {
    const float* w1 = FIN(Lw1[l]);
    // A = h @ w1[:64] + b1 ; BmT = (h @ w1[64:128])^T
    launch_dense(h, w1, FIN(Lb1[l]), nullptr, Amat, NNODES, HDIM, EHID, EHID, 0, 0, stream);
    launch_dense(h, w1 + 64 * EHID, nullptr, nullptr, BmT, NNODES, HDIM, EHID, EHID, 0, 1, stream);
    egnn_edge_layer_kernel<<<NNODES / WAVES, WAVES * 32, lds1, stream>>>(
        xin, Amat, BmT, w1 + 128 * EHID, adj,
        FIN(Lw2[l]), FIN(Lb2[l]), FIN(Lcw1[l]), FIN(Lcb1[l]), FIN(Lcw2[l]), FIN(Lcb2[l]),
        mi, xouts[l]);
    // node update: h += silu([h, m_i] @ nw1 + nb1) @ nw2 + nb2
    concat_kernel<<<(NNODES * 80 + 255) / 256, 256, 0, stream>>>(h, mi, hc);
    launch_dense(hc, FIN(Lnw1[l]), FIN(Lnb1[l]), nullptr, t1, NNODES, 80, 128, 128, 1, 0, stream);
    launch_dense(t1, FIN(Lnw2[l]), FIN(Lnb2[l]), h, h, NNODES, 128, HDIM, HDIM, 0, 0, stream);
    xin = xouts[l];
  }

  // final stage: P = h @ ew1[:64] + eb1 ; QT = (h @ ew1[64:128])^T
  const float* ew1 = FIN(I_ew1);
  launch_dense(h, ew1, FIN(I_eb1), nullptr, Pbuf, NNODES, HDIM, HDIM, HDIM, 0, 0, stream);
  launch_dense(h, ew1 + 64 * HDIM, nullptr, nullptr, QT, NNODES, HDIM, HDIM, HDIM, 0, 1, stream);
  egnn_final_edge_kernel<<<NNODES / WAVES, WAVES * 32, lds2, stream>>>(
      coors, Pbuf, QT, ew1 + 128 * HDIM, adj,
      FIN(I_ew2), FIN(I_eb2), FIN(I_xw1), FIN(I_xb1), FIN(I_xw2), FIN(I_xb2), nbr, nnb);

  // gamma = sigmoid(relu(h @ vw1 + vb1) @ vw2 + vb2)
  launch_dense(h, FIN(I_vw1), FIN(I_vb1), nullptr, tv, NNODES, HDIM, HDIM, HDIM, 2, 0, stream);
  launch_dense(tv, FIN(I_vw2), FIN(I_vb2), nullptr, gbuf, NNODES, HDIM, 1, 1, 3, 0, stream);
  // stress = h @ sw + sb
  launch_dense(h, FIN(I_sw), FIN(I_sb), nullptr, out_stress, NNODES, HDIM, 1, 1, 0, 0, stream);
  // combine outputs
  combine_kernel<<<(NNODES * 3 + 255) / 256, 256, 0, stream>>>(
      feats, coors, gbuf, nbr, nnb, FIN(I_C), out_vel, out_coors);
  #undef FIN
}
